// Casual_11948599017518
// MI455X (gfx1250) — matrix operand order repaired and checked by reading, not compiled, not run on hardware
//
#include <hip/hip_runtime.h>

// Fused MoE for gfx1250 (CDNA5, wave32) with v_wmma_f32_16x16x32_bf16.
// Shapes: IN=256, HID=512, EOUT=256, E=8, OUT=256, T=32768.
//
// Pass 1 (prep): convert w1/w2/wo (f32) into bf16 in d_ws, pre-swizzled into
//   B-fragment order: per K=32 chunk, [column n][k 0..31] contiguous (64B/col).
// Pass 2 (main): per-WG 32 tokens; fc1/fc2 per expert with gate-weighted
//   accumulation in registers; final projection. B fragments load straight
//   from global (L2-resident bf16), A fragments from LDS.

typedef __attribute__((ext_vector_type(16))) __bf16 v16bf;
typedef __attribute__((ext_vector_type(8)))  float  v8f;

#define TOK   32
#define INC   256
#define HID   512
#define EOUTC 256
#define NEXP  8
#define OUTC  256

// padded LDS row strides (elements)
#define XLD   264   // 256 + 8
#define HLD   520   // 512 + 8
#define ALD   264

// d_ws layout (bf16 elements)
#define W1_ELEMS (NEXP * (INC/32) * HID * 32)    // 1048576
#define W2_ELEMS (NEXP * (HID/32) * EOUTC * 32)  // 1048576
#define WO_ELEMS ((EOUTC/32) * OUTC * 32)        // 65536
#define NCOL_W1  (NEXP * (INC/32) * HID)         // 32768 columns
#define NCOL_W2  (NEXP * (HID/32) * EOUTC)       // 32768
#define NCOL_WO  ((EOUTC/32) * OUTC)             // 2048

union FragU { uint4 q[2]; v16bf v; };
union Pack4 { __bf16 b[4]; uint2 u2; };
union Col32 { __bf16 b[32]; uint4 q[4]; };

// A fragment: 16x32 bf16 (MxK) from row-major LDS (stride lda elements).
__device__ __forceinline__ v16bf load_a(const __bf16* base, int row0,
                                        int lda, int kc, int lane) {
  int m  = lane & 15;
  int ko = (lane >> 4) << 3;              // 0 or 8
  const __bf16* p = base + (row0 + m) * lda + kc + ko;
  FragU u;
  u.q[0] = *(const uint4*)(p);            // K = ko .. ko+7
  u.q[1] = *(const uint4*)(p + 16);       // K = ko+16 .. ko+23
  return u.v;
}

// B fragment: 32x16 bf16 (KxN) straight from global in column-chunk layout:
// chunk base points at [n][32]; element (k,n) at base[n*32 + k].
__device__ __forceinline__ v16bf load_b(const __bf16* chunk, int n0, int lane) {
  int n  = lane & 15;
  int kb = (lane >> 4) << 3;              // 0 or 8
  const __bf16* p = chunk + (size_t)(n0 + n) * 32 + kb;
  FragU u;
  u.q[0] = *(const uint4*)(p);
  u.q[1] = *(const uint4*)(p + 16);
  return u.v;
}

__device__ __forceinline__ v8f wmma_bf16(v16bf a, v16bf b, v8f c) {
  return __builtin_amdgcn_wmma_f32_16x16x32_bf16(false, a, false, b,
                                                 (short)0, c, false, false);
}

// ---------------- pass 1: weight conversion / swizzle ----------------
__global__ __launch_bounds__(256) void moe_prep_weights(
    const float* __restrict__ w1, const float* __restrict__ w2,
    const float* __restrict__ wo, __bf16* __restrict__ ws) {
  int col = blockIdx.x * blockDim.x + threadIdx.x;
  __bf16* w1b = ws;
  __bf16* w2b = ws + W1_ELEMS;
  __bf16* wob = w2b + W2_ELEMS;

  Col32 c32;
  __bf16* dst;
  if (col < NCOL_W1) {
    int n = col & (HID - 1);
    int c = (col >> 9) & 7;               // K-chunk
    int e = col >> 12;
    const float* src = w1 + ((size_t)e * INC + c * 32) * HID + n;
#pragma unroll
    for (int k = 0; k < 32; ++k) c32.b[k] = (__bf16)src[(size_t)k * HID];
    dst = w1b + (size_t)col * 32;
  } else if (col < NCOL_W1 + NCOL_W2) {
    int cc = col - NCOL_W1;
    int n = cc & (EOUTC - 1);
    int c = (cc >> 8) & 15;
    int e = cc >> 12;
    const float* src = w2 + ((size_t)e * HID + c * 32) * EOUTC + n;
#pragma unroll
    for (int k = 0; k < 32; ++k) c32.b[k] = (__bf16)src[(size_t)k * EOUTC];
    dst = w2b + (size_t)cc * 32;
  } else if (col < NCOL_W1 + NCOL_W2 + NCOL_WO) {
    int cc = col - (NCOL_W1 + NCOL_W2);
    int n = cc & (OUTC - 1);
    int c = cc >> 8;
    const float* src = wo + ((size_t)c * 32) * OUTC + n;
#pragma unroll
    for (int k = 0; k < 32; ++k) c32.b[k] = (__bf16)src[(size_t)k * OUTC];
    dst = wob + (size_t)cc * 32;
  } else {
    return;
  }
  uint4* d4 = (uint4*)dst;
#pragma unroll
  for (int i = 0; i < 4; ++i) d4[i] = c32.q[i];
}

// ---------------- pass 2: fused MoE ----------------
__global__ __launch_bounds__(256) void moe_fused_wmma(
    const float* __restrict__ x,  const float* __restrict__ b1,
    const float* __restrict__ b2, const float* __restrict__ wg,
    const float* __restrict__ bg, const float* __restrict__ bo,
    const __bf16* __restrict__ ws, float* __restrict__ out) {

  __shared__ __align__(16) __bf16 xA[TOK * XLD];   // x tile (bf16, row-major)
  __shared__ __align__(16) __bf16 hA[TOK * HLD];   // hidden (bf16, row-major)
  __shared__ __align__(16) __bf16 aW[TOK * ALD];   // weighted sum (bf16)
  __shared__ float gateS[TOK * NEXP];

  const __bf16* w1b = ws;
  const __bf16* w2b = ws + W1_ELEMS;
  const __bf16* wob = w2b + W2_ELEMS;

  const int tid  = threadIdx.x;
  const int lane = tid & 31;
  const int wv   = tid >> 5;
  const long t0  = (long)blockIdx.x * TOK;

  // ---- stage x tile [32 x 256] f32 -> bf16 (packed 8B stores) ----
#pragma unroll
  for (int it = 0; it < 8; ++it) {
    int idx = it * 256 + tid;              // float4 index, 64 per row
    int r   = idx >> 6;
    int c4  = (idx & 63) << 2;
    float4 v = *(const float4*)(x + (t0 + r) * INC + c4);
    Pack4 p;
    p.b[0] = (__bf16)v.x; p.b[1] = (__bf16)v.y;
    p.b[2] = (__bf16)v.z; p.b[3] = (__bf16)v.w;
    *(uint2*)(&xA[r * XLD + c4]) = p.u2;
  }
  __syncthreads();

  // ---- gating: one (t,e) pair per thread, then per-token softmax ----
  {
    int t = tid >> 3, e = tid & 7;
    float acc = bg[e];
    for (int c = 0; c < INC; ++c)
      acc += (float)xA[t * XLD + c] * wg[c * NEXP + e];
    gateS[t * NEXP + e] = acc;
  }
  __syncthreads();
  if (tid < TOK) {
    float m = -3.0e38f;
#pragma unroll
    for (int e = 0; e < NEXP; ++e) m = fmaxf(m, gateS[tid * NEXP + e]);
    float ex[NEXP], s = 0.f;
#pragma unroll
    for (int e = 0; e < NEXP; ++e) { ex[e] = __expf(gateS[tid * NEXP + e] - m); s += ex[e]; }
    float inv = 1.0f / s;
#pragma unroll
    for (int e = 0; e < NEXP; ++e) gateS[tid * NEXP + e] = ex[e] * inv;
  }
  __syncthreads();

  v8f wacc[2][2];
#pragma unroll
  for (int mt = 0; mt < 2; ++mt)
#pragma unroll
    for (int j = 0; j < 2; ++j) wacc[mt][j] = (v8f){0,0,0,0,0,0,0,0};

  for (int e = 0; e < NEXP; ++e) {
    // ================= fc1: [32x256] x [256x512] =================
    v8f acc1[2][4];
#pragma unroll
    for (int mt = 0; mt < 2; ++mt)
#pragma unroll
      for (int j = 0; j < 4; ++j) acc1[mt][j] = (v8f){0,0,0,0,0,0,0,0};

#pragma unroll 2
    for (int c = 0; c < INC / 32; ++c) {
      const __bf16* chunk = w1b + ((size_t)(e * (INC / 32) + c) * HID) * 32;
      v16bf a0 = load_a(xA,  0, XLD, c * 32, lane);
      v16bf a1 = load_a(xA, 16, XLD, c * 32, lane);
#pragma unroll
      for (int j = 0; j < 4; ++j) {
        int n0 = (wv + j * 8) * 16;
        v16bf bf = load_b(chunk, n0, lane);
        acc1[0][j] = wmma_bf16(a0, bf, acc1[0][j]);
        acc1[1][j] = wmma_bf16(a1, bf, acc1[1][j]);
      }
    }
    __syncthreads();    // all waves done reading hA (prev expert's fc2)
    // epilogue: bias + relu -> hA (bf16)
#pragma unroll
    for (int mt = 0; mt < 2; ++mt)
#pragma unroll
      for (int j = 0; j < 4; ++j) {
        int n = (wv + j * 8) * 16 + (lane & 15);
        float bias = b1[e * HID + n];
#pragma unroll
        for (int i = 0; i < 8; ++i) {
          int t = mt * 16 + i + ((lane >> 4) << 3);
          float hv = acc1[mt][j][i] + bias;
          hA[t * HLD + n] = (__bf16)(hv > 0.f ? hv : 0.f);
        }
      }
    __syncthreads();    // hA visible to all waves

    // ================= fc2: [32x512] x [512x256] =================
    v8f acc2[2][2];
#pragma unroll
    for (int mt = 0; mt < 2; ++mt)
#pragma unroll
      for (int j = 0; j < 2; ++j) acc2[mt][j] = (v8f){0,0,0,0,0,0,0,0};

#pragma unroll 2
    for (int c = 0; c < HID / 32; ++c) {
      const __bf16* chunk = w2b + ((size_t)(e * (HID / 32) + c) * EOUTC) * 32;
      v16bf a0 = load_a(hA,  0, HLD, c * 32, lane);
      v16bf a1 = load_a(hA, 16, HLD, c * 32, lane);
#pragma unroll
      for (int j = 0; j < 2; ++j) {
        int n0 = (wv + j * 8) * 16;
        v16bf bf = load_b(chunk, n0, lane);
        acc2[0][j] = wmma_bf16(a0, bf, acc2[0][j]);
        acc2[1][j] = wmma_bf16(a1, bf, acc2[1][j]);
      }
    }
    // gate-weighted accumulate in registers
#pragma unroll
    for (int mt = 0; mt < 2; ++mt)
#pragma unroll
      for (int j = 0; j < 2; ++j) {
        int n = (wv + j * 8) * 16 + (lane & 15);
        float bias = b2[e * EOUTC + n];
#pragma unroll
        for (int i = 0; i < 8; ++i) {
          int t = mt * 16 + i + ((lane >> 4) << 3);
          float g = gateS[t * NEXP + e];
          wacc[mt][j][i] += g * (acc2[mt][j][i] + bias);
        }
      }
  } // experts

  // ---- weighted sum -> LDS as bf16 A-matrix ----
#pragma unroll
  for (int mt = 0; mt < 2; ++mt)
#pragma unroll
    for (int j = 0; j < 2; ++j) {
      int n = (wv + j * 8) * 16 + (lane & 15);
#pragma unroll
      for (int i = 0; i < 8; ++i) {
        int t = mt * 16 + i + ((lane >> 4) << 3);
        aW[t * ALD + n] = (__bf16)wacc[mt][j][i];
      }
    }
  __syncthreads();

  // ================= final: [32x256] x [256x256] =================
  v8f acc3[2][2];
#pragma unroll
  for (int mt = 0; mt < 2; ++mt)
#pragma unroll
    for (int j = 0; j < 2; ++j) acc3[mt][j] = (v8f){0,0,0,0,0,0,0,0};

#pragma unroll 2
  for (int c = 0; c < EOUTC / 32; ++c) {
    const __bf16* chunk = wob + ((size_t)c * OUTC) * 32;
    v16bf a0 = load_a(aW,  0, ALD, c * 32, lane);
    v16bf a1 = load_a(aW, 16, ALD, c * 32, lane);
#pragma unroll
    for (int j = 0; j < 2; ++j) {
      int n0 = (wv + j * 8) * 16;
      v16bf bf = load_b(chunk, n0, lane);
      acc3[0][j] = wmma_bf16(a0, bf, acc3[0][j]);
      acc3[1][j] = wmma_bf16(a1, bf, acc3[1][j]);
    }
  }

  // ---- bias + store out (f32) ----
#pragma unroll
  for (int mt = 0; mt < 2; ++mt)
#pragma unroll
    for (int j = 0; j < 2; ++j) {
      int n = (wv + j * 8) * 16 + (lane & 15);
      float bias = bo[n];
#pragma unroll
      for (int i = 0; i < 8; ++i) {
        int t = mt * 16 + i + ((lane >> 4) << 3);
        out[(t0 + t) * OUTC + n] = acc3[mt][j][i] + bias;
      }
    }
}

extern "C" void kernel_launch(void* const* d_in, const int* in_sizes, int n_in,
                              void* d_out, int out_size, void* d_ws, size_t ws_size,
                              hipStream_t stream) {
  const float* x  = (const float*)d_in[0];
  const float* w1 = (const float*)d_in[1];
  const float* b1 = (const float*)d_in[2];
  const float* w2 = (const float*)d_in[3];
  const float* b2 = (const float*)d_in[4];
  const float* wg = (const float*)d_in[5];
  const float* bg = (const float*)d_in[6];
  const float* wo = (const float*)d_in[7];
  const float* bo = (const float*)d_in[8];
  float* out = (float*)d_out;
  __bf16* ws = (__bf16*)d_ws;

  // pass 1: weight convert + swizzle (67584 columns)
  int ncol = NCOL_W1 + NCOL_W2 + NCOL_WO;
  moe_prep_weights<<<(ncol + 255) / 256, 256, 0, stream>>>(w1, w2, wo, ws);

  // pass 2: fused MoE
  int T = in_sizes[0] / INC;               // 32768 tokens
  moe_fused_wmma<<<T / TOK, 256, 0, stream>>>(x, b1, b2, wg, bg, bo, ws, out);
}
